// LinAngularAttention_26688926777397
// MI455X (gfx1250) — matrix operand order repaired and checked by reading, not compiled, not run on hardware
//
#include <hip/hip_runtime.h>
#include <stdint.h>

// ---------------------------------------------------------------------------
// LinAngularAttention for MI455X (gfx1250, wave32, WMMA bf16-split, TDM tiles,
// double-buffered async pipeline).  B=4 L=4096 C=1024 H=16 D=64 KS=9
// ---------------------------------------------------------------------------

#define B_   4
#define L_   4096
#define C_   1024
#define H_   16
#define D_   64
#define KS_  9
#define INV_PI 0.31830988618379067f

typedef __attribute__((ext_vector_type(16))) __bf16 v16bf;
typedef __attribute__((ext_vector_type(8)))  float  v8f;
typedef __attribute__((ext_vector_type(4)))  unsigned int v4u;
typedef __attribute__((ext_vector_type(8)))  int    v8i;
typedef __attribute__((ext_vector_type(4)))  int    v4i;
typedef __attribute__((ext_vector_type(2)))  float  f32x2;

// ---- TDM availability probe -----------------------------------------------
#if defined(__has_builtin)
#if __has_builtin(__builtin_amdgcn_tensor_load_to_lds)
#define HAVE_TDM 1
#endif
#endif
#ifndef HAVE_TDM
#define HAVE_TDM 0
#endif

// ---- bf16 split: x ~= hi + lo (both bf16), packed pairwise into u32 -------
__device__ __forceinline__ void pack_split2(float a, float b,
                                            unsigned& hi, unsigned& lo) {
  __bf16 ha = (__bf16)a, hb = (__bf16)b;          // native RNE convert
  float fa = (float)ha, fb = (float)hb;
  __bf16 la = (__bf16)(a - fa), lb = (__bf16)(b - fb);
  unsigned short sa, sb, ta, tb;
  __builtin_memcpy(&sa, &ha, 2); __builtin_memcpy(&sb, &hb, 2);
  __builtin_memcpy(&ta, &la, 2); __builtin_memcpy(&tb, &lb, 2);
  hi = (unsigned)sa | ((unsigned)sb << 16);
  lo = (unsigned)ta | ((unsigned)tb << 16);
}

union FragU {              // 16 bf16 = 8 u32 = 2 x b128
  v16bf v;
  unsigned u[8];
  v4u q[2];
};

// Load one WMMA fragment from a u32-packed bf16 plane: two b128 LDS reads.
__device__ __forceinline__ v16bf load_frag(const unsigned* plane, int base, int step) {
  FragU f;
  f.q[0] = *(const v4u*)(plane + base);
  f.q[1] = *(const v4u*)(plane + base + step);
  return f.v;
}

__device__ __forceinline__ v8f wmma_bf16(v16bf a, v16bf b, v8f c) {
  return __builtin_amdgcn_wmma_f32_16x16x32_bf16(false, a, false, b,
                                                 (short)0, c, false, false);
}

__device__ __forceinline__ v8f vzero8() {
  v8f z = {0.f, 0.f, 0.f, 0.f, 0.f, 0.f, 0.f, 0.f};
  return z;
}

// ---- Tensor Data Mover: 2-D tile of 4-byte elements, global -> LDS --------
#if HAVE_TDM
__device__ __forceinline__ void tdm_load_2d(const void* gptr, unsigned lds_off,
                                            unsigned tile_x, unsigned tile_y,
                                            unsigned row_stride_elems) {
  unsigned long long ga = (unsigned long long)gptr;
  v4u g0;
  g0[0] = 1u;                                                // count=1
  g0[1] = lds_off;                                           // lds_addr
  g0[2] = (unsigned)ga;                                      // addr[31:0]
  g0[3] = (unsigned)((ga >> 32) & 0x01FFFFFFu) | 0x80000000u;// addr[56:32],type=2
  v8i g1;
  g1[0] = (int)(2u << 16);                                   // data_size=4B
  g1[1] = (int)((tile_x & 0xFFFFu) << 16);                   // tensor_dim0 lo
  g1[2] = (int)((tile_x >> 16) | ((tile_y & 0xFFFFu) << 16));// dim0 hi|dim1 lo
  g1[3] = (int)((tile_y >> 16) | (tile_x << 16));            // dim1 hi|tile_dim0
  g1[4] = (int)(tile_y & 0xFFFFu);                           // tile_dim1
  g1[5] = (int)row_stride_elems;                             // dim0_stride lo32
  g1[6] = 0;
  g1[7] = 0;
  v4i z4 = {0, 0, 0, 0};
#if __clang_major__ >= 23
  v8i z8 = {0, 0, 0, 0, 0, 0, 0, 0};
  __builtin_amdgcn_tensor_load_to_lds(g0, g1, z4, z4, z8, 0);
#else
  __builtin_amdgcn_tensor_load_to_lds(g0, g1, z4, z4, 0);
#endif
}
#endif

// Stage a TY x TX tile of 4-byte elements (row stride rs) into packed LDS.
template <int TX, int TY, typename T>
__device__ __forceinline__ void stage_tile(T* lds, const T* g, int rs,
                                           int tid, int nthr) {
  static_assert(sizeof(T) == 4, "4-byte elements only");
#if HAVE_TDM
  if (tid < 32) {
    tdm_load_2d(g, (unsigned)(unsigned long long)lds,
                (unsigned)TX, (unsigned)TY, (unsigned)rs);
  }
  (void)nthr;
#else
  for (int idx = tid; idx < TX * TY; idx += nthr) {
    int r = idx / TX, c = idx - r * TX;
    lds[idx] = g[(size_t)r * rs + c];
  }
#endif
}

// Wave 0 waits for all in-flight TDM transfers (its private TENSORcnt).
__device__ __forceinline__ void tdm_drain(int tid) {
#if HAVE_TDM
  if (tid < 32) __builtin_amdgcn_s_wait_tensorcnt(0);
#endif
  (void)tid;
}

// ---------------------------------------------------------------------------
// Weight prep: W(KxN f32) -> transposed u32-packed bf16 planes [n][K/2]
// (exactly the B-fragment plane layout, so GEMM B needs no conversion).
// ---------------------------------------------------------------------------
__global__ __launch_bounds__(256) void prep_weight_planes(
    const float* __restrict__ W, unsigned* __restrict__ hiP,
    unsigned* __restrict__ loP, int K, int N)
{
  int idx = blockIdx.x * blockDim.x + threadIdx.x;   // over N*(K/2)
  int total = N * (K >> 1);
  if (idx >= total) return;
  int j = idx / N;              // k-pair index (n fastest => coalesced reads)
  int n = idx - j * N;
  float a = W[(size_t)(2 * j) * N + n];
  float b = W[(size_t)(2 * j + 1) * N + n];
  unsigned hi, lo;
  pack_split2(a, b, hi, lo);
  hiP[(size_t)n * (K >> 1) + j] = hi;
  loP[(size_t)n * (K >> 1) + j] = lo;
}

// ---------------------------------------------------------------------------
// GEMM: Out(16384xN) = A(16384xK) @ W(KxN) [+bias] [L2-norm aligned 64-groups]
// Block 128x64, 8 waves; double-buffered TDM pipeline: tile i+1 DMA overlaps
// split + 12 WMMAs of tile i.
// ---------------------------------------------------------------------------
#define BM 128
#define BN 64
#define BK 32

__global__ __launch_bounds__(256) void gemm_bf16split(
    const float* __restrict__ A,
    const unsigned* __restrict__ WHi, const unsigned* __restrict__ WLo,
    float* __restrict__ Out, const float* __restrict__ bias,
    int N, int K, int do_norm)
{
  __shared__ __attribute__((aligned(16))) float    lA [2][BM * BK];
  __shared__ __attribute__((aligned(16))) unsigned bHi[2][BN * (BK / 2)];
  __shared__ __attribute__((aligned(16))) unsigned bLo[2][BN * (BK / 2)];
  __shared__ __attribute__((aligned(16))) unsigned aHi[BM * (BK / 2)];
  __shared__ __attribute__((aligned(16))) unsigned aLo[BM * (BK / 2)];

  const int tid  = threadIdx.x;
  const int wave = tid >> 5;
  const int lane = tid & 31;
  const int nl   = lane & 15;
  const int hf   = lane >> 4;
  const int m0   = blockIdx.x * BM;
  const int n0   = blockIdx.y * BN;
  const int Kh   = K >> 1;                   // weight plane row length (u32)

  auto issue = [&](int buf, int kc) {
    stage_tile<BK, BM>(lA[buf], A + (size_t)m0 * K + kc, K, tid, 256);
    stage_tile<BK / 2, BN>(bHi[buf], WHi + (size_t)n0 * Kh + (kc >> 1), Kh, tid, 256);
    stage_tile<BK / 2, BN>(bLo[buf], WLo + (size_t)n0 * Kh + (kc >> 1), Kh, tid, 256);
  };

  v8f acc[4];
#pragma unroll
  for (int t = 0; t < 4; ++t) acc[t] = vzero8();

  issue(0, 0);                               // prologue: tile 0 in flight
  int cur = 0;
  for (int kc = 0; kc < K; kc += BK) {
    tdm_drain(tid);                          // tile(kc) landed (wave 0)
    __syncthreads();                         // publish tile; prior reads done
    if (kc + BK < K) issue(cur ^ 1, kc + BK);// overlap next DMA with compute

    // Split A tile once (cooperative): raw f32 -> hi/lo planes [m][k/2]
    const float* la = lA[cur];
    for (int idx = tid; idx < BM * (BK / 2); idx += 256) {
      int m = idx >> 4, j = idx & 15;
      unsigned hi, lo;
      pack_split2(la[m * BK + 2 * j], la[m * BK + 2 * j + 1], hi, lo);
      aHi[idx] = hi;
      aLo[idx] = lo;
    }
    __syncthreads();                         // planes ready

    const int abase = (16 * wave + nl) * 16 + hf * 4;
    v16bf ah = load_frag(aHi, abase, 8);
    v16bf al = load_frag(aLo, abase, 8);
#pragma unroll
    for (int t = 0; t < 4; ++t) {
      const int bbase = (t * 16 + nl) * 16 + hf * 4;
      v16bf Bh = load_frag(bHi[cur], bbase, 8);
      v16bf Bl = load_frag(bLo[cur], bbase, 8);
      acc[t] = wmma_bf16(ah, Bh, acc[t]);
      acc[t] = wmma_bf16(ah, Bl, acc[t]);
      acc[t] = wmma_bf16(al, Bh, acc[t]);
    }
    cur ^= 1;
  }

  // Epilogue: optional per-row L2 norm over this 64-col tile, bias, store.
#pragma unroll
  for (int r = 0; r < 8; ++r) {
    if (do_norm) {
      float ss = 0.f;
#pragma unroll
      for (int t = 0; t < 4; ++t) ss += acc[t][r] * acc[t][r];
      ss += __shfl_xor(ss, 1, 32);
      ss += __shfl_xor(ss, 2, 32);
      ss += __shfl_xor(ss, 4, 32);
      ss += __shfl_xor(ss, 8, 32);
      float inv = 1.f / sqrtf(ss);
#pragma unroll
      for (int t = 0; t < 4; ++t) acc[t][r] = acc[t][r] * inv;
    }
    const int m = m0 + 16 * wave + 8 * hf + r;
#pragma unroll
    for (int t = 0; t < 4; ++t) {
      const int n = n0 + 16 * t + nl;
      float vv = acc[t][r];
      if (bias) vv += bias[n];
      Out[(size_t)m * N + n] = vv;
    }
  }
}

// ---------------------------------------------------------------------------
// attn per (b,h): attn(64x64) = kn^T @ vh (K = L = 4096), double-buffered.
// Output written as u32-packed bf16 hi/lo planes [e][d/2] (combine B layout).
// ---------------------------------------------------------------------------
__global__ __launch_bounds__(128) void attn_kernel(
    const float* __restrict__ kn, const float* __restrict__ vh,
    unsigned* __restrict__ attnHi, unsigned* __restrict__ attnLo)
{
  __shared__ __attribute__((aligned(16))) float    lK [2][32 * D_];
  __shared__ __attribute__((aligned(16))) float    lV [2][32 * D_];
  __shared__ __attribute__((aligned(16))) unsigned kHi[D_ * 16];  // [d][l/2]
  __shared__ __attribute__((aligned(16))) unsigned kLo[D_ * 16];
  __shared__ __attribute__((aligned(16))) unsigned vHi[D_ * 16];  // [e][l/2]
  __shared__ __attribute__((aligned(16))) unsigned vLo[D_ * 16];

  const int tid  = threadIdx.x;
  const int wave = tid >> 5;
  const int lane = tid & 31;
  const int nl = lane & 15, hf = lane >> 4;
  const int bh = blockIdx.x;
  const float* kb = kn + (size_t)bh * (L_ * D_);
  const float* vb = vh + (size_t)bh * (L_ * D_);

  auto issue = [&](int buf, int lc) {
    stage_tile<D_, 32>(lK[buf], kb + (size_t)lc * D_, D_, tid, 128);
    stage_tile<D_, 32>(lV[buf], vb + (size_t)lc * D_, D_, tid, 128);
  };

  v8f acc[4];
#pragma unroll
  for (int t = 0; t < 4; ++t) acc[t] = vzero8();

  issue(0, 0);
  int cur = 0;
  for (int lc = 0; lc < L_; lc += 32) {
    tdm_drain(tid);
    __syncthreads();
    if (lc + 32 < L_) issue(cur ^ 1, lc + 32);

    // Transpose + split once: raw [l][d] f32 -> planes [d][l/2] u32
    const float* lk = lK[cur];
    const float* lv = lV[cur];
    for (int idx = tid; idx < D_ * 16; idx += 128) {
      int d = idx >> 4, j = idx & 15, l = 2 * j;
      unsigned hi, lo;
      pack_split2(lk[l * D_ + d], lk[(l + 1) * D_ + d], hi, lo);
      kHi[idx] = hi; kLo[idx] = lo;
      pack_split2(lv[l * D_ + d], lv[(l + 1) * D_ + d], hi, lo);
      vHi[idx] = hi; vLo[idx] = lo;
    }
    __syncthreads();

    const int abase = (16 * wave + nl) * 16 + hf * 4;   // A = kn^T row d
    v16bf ah = load_frag(kHi, abase, 8);
    v16bf al = load_frag(kLo, abase, 8);
#pragma unroll
    for (int t = 0; t < 4; ++t) {
      const int bbase = (t * 16 + nl) * 16 + hf * 4;    // B col e
      v16bf Bh = load_frag(vHi, bbase, 8);
      v16bf Bl = load_frag(vLo, bbase, 8);
      acc[t] = wmma_bf16(ah, Bh, acc[t]);
      acc[t] = wmma_bf16(ah, Bl, acc[t]);
      acc[t] = wmma_bf16(al, Bh, acc[t]);
    }
    cur ^= 1;
  }

  // Store as hi/lo planes [e][d/2]: pairs (d, d+1) live in adjacent acc rows.
#pragma unroll
  for (int t = 0; t < 4; ++t) {
    const int e = t * 16 + nl;
#pragma unroll
    for (int r = 0; r < 8; r += 2) {
      const int d = 16 * wave + 8 * hf + r;
      unsigned hi, lo;
      pack_split2(acc[t][r], acc[t][r + 1], hi, lo);
      attnHi[(size_t)bh * (D_ * 32) + e * 32 + (d >> 1)] = hi;
      attnLo[(size_t)bh * (D_ * 32) + e * 32 + (d >> 1)] = lo;
    }
  }
}

// ---------------------------------------------------------------------------
// combine per (b,h), 128-row strip:
//   x = 0.5*vh + (1/pi)*(qn @ attn); x /= ||x||_row; y = x + depthwise_conv(vh)
// ---------------------------------------------------------------------------
__global__ __launch_bounds__(256) void combine_kernel(
    const float* __restrict__ qn, const float* __restrict__ vh,
    const unsigned* __restrict__ attnHi, const unsigned* __restrict__ attnLo,
    const float* __restrict__ wdconv, float* __restrict__ y)
{
  __shared__ __attribute__((aligned(16))) unsigned atHi[D_ * 32];  // [e][d/2]
  __shared__ __attribute__((aligned(16))) unsigned atLo[D_ * 32];

  const int tid = threadIdx.x, wave = tid >> 5, lane = tid & 31;
  const int nl = lane & 15, hf = lane >> 4;
  const int bh = blockIdx.y, h = bh & (H_ - 1);
  const int l0 = blockIdx.x * 128;
  const float* qb = qn + (size_t)bh * (L_ * D_);
  const float* vb = vh + (size_t)bh * (L_ * D_);
  float* yb = y + (size_t)bh * (L_ * D_);

  stage_tile<32, D_>(atHi, attnHi + (size_t)bh * (D_ * 32), 32, tid, 256);
  stage_tile<32, D_>(atLo, attnLo + (size_t)bh * (D_ * 32), 32, tid, 256);
  tdm_drain(tid);
  __syncthreads();

  float wd[KS_];
#pragma unroll
  for (int j = 0; j < KS_; ++j) wd[j] = wdconv[h * KS_ + j];

  v8f acc[4];
#pragma unroll
  for (int t = 0; t < 4; ++t) acc[t] = vzero8();

  const int lrow = l0 + 16 * wave + nl;
#pragma unroll
  for (int kc = 0; kc < D_; kc += 32) {
    // A fragment from global qn: 2 runs of 4 contiguous f32 pairs -> split in registers
    const f32x2* qp = (const f32x2*)(qb + (size_t)lrow * D_ + kc);
    FragU ah, al;
#pragma unroll
    for (int j = 0; j < 8; ++j) {
      const int pidx = (j < 4) ? (hf * 4 + j) : (8 + hf * 4 + (j - 4));
      f32x2 p = qp[pidx];
      pack_split2(p.x, p.y, ah.u[j], al.u[j]);
    }
#pragma unroll
    for (int t = 0; t < 4; ++t) {
      const int bbase = (t * 16 + nl) * 32 + (kc >> 1) + hf * 4;
      v16bf Bh = load_frag(atHi, bbase, 8);
      v16bf Bl = load_frag(atLo, bbase, 8);
      acc[t] = wmma_bf16(ah.v, Bh, acc[t]);
      acc[t] = wmma_bf16(ah.v, Bl, acc[t]);
      acc[t] = wmma_bf16(al.v, Bh, acc[t]);
    }
  }

#pragma unroll
  for (int r = 0; r < 8; ++r) {
    const int lr = l0 + 16 * wave + 8 * hf + r;
    float xv[4];
    float ss = 0.f;
#pragma unroll
    for (int t = 0; t < 4; ++t) {
      const int col = t * 16 + nl;
      float vc = vb[(size_t)lr * D_ + col];
      xv[t] = 0.5f * vc + INV_PI * acc[t][r];
      ss += xv[t] * xv[t];
    }
    ss += __shfl_xor(ss, 1, 32);
    ss += __shfl_xor(ss, 2, 32);
    ss += __shfl_xor(ss, 4, 32);
    ss += __shfl_xor(ss, 8, 32);
    const float inv = 1.f / sqrtf(ss);
#pragma unroll
    for (int t = 0; t < 4; ++t) {
      const int col = t * 16 + nl;
      float conv = 0.f;
#pragma unroll
      for (int j = 0; j < KS_; ++j) {
        const int ls = lr + j - (KS_ / 2);
        float s = (ls >= 0 && ls < L_) ? vb[(size_t)ls * D_ + col] : 0.f;
        conv += wd[j] * s;
      }
      yb[(size_t)lr * D_ + col] = xv[t] * inv + conv;
    }
  }
}

// ---------------------------------------------------------------------------
// Launcher. Inputs: q,k,v,Wq,Wk,Wv,Wproj,bproj,Wdconv (all f32).
// ws: qn | kn(->y) | vh (f32, 64MB each) | attn planes | 4x weight planes.
// ---------------------------------------------------------------------------
extern "C" void kernel_launch(void* const* d_in, const int* in_sizes, int n_in,
                              void* d_out, int out_size, void* d_ws, size_t ws_size,
                              hipStream_t stream) {
  (void)in_sizes; (void)n_in; (void)out_size; (void)ws_size;
  const float* q     = (const float*)d_in[0];
  const float* k     = (const float*)d_in[1];
  const float* v     = (const float*)d_in[2];
  const float* Wq    = (const float*)d_in[3];
  const float* Wk    = (const float*)d_in[4];
  const float* Wv    = (const float*)d_in[5];
  const float* Wproj = (const float*)d_in[6];
  const float* bproj = (const float*)d_in[7];
  const float* Wd    = (const float*)d_in[8];
  float* out = (float*)d_out;

  float* ws = (float*)d_ws;
  const size_t MN = (size_t)(B_ * L_) * C_;            // 16,777,216
  float* qn = ws;
  float* kn = ws + MN;
  float* vh = ws + 2 * MN;
  float* y  = kn;                                      // kn dead after attn

  unsigned* u0 = (unsigned*)(ws + 3 * MN);
  const size_t APLANE = (size_t)(B_ * H_) * D_ * 32;   // 131,072 u32
  unsigned* attnHi = u0;
  unsigned* attnLo = u0 + APLANE;
  const size_t WPLANE = (size_t)C_ * (C_ / 2);         // 524,288 u32
  unsigned* wqHi = attnLo + APLANE;
  unsigned* wqLo = wqHi + WPLANE;
  unsigned* wkHi = wqLo + WPLANE;
  unsigned* wkLo = wkHi + WPLANE;
  unsigned* wvHi = wkLo + WPLANE;
  unsigned* wvLo = wvHi + WPLANE;
  unsigned* wpHi = wvLo + WPLANE;
  unsigned* wpLo = wpHi + WPLANE;

  // 1) Pre-split weights into transposed bf16 hi/lo planes (once per call).
  const int prep_blocks = (int)((WPLANE + 255) / 256);
  prep_weight_planes<<<prep_blocks, 256, 0, stream>>>(Wq, wqHi, wqLo, C_, C_);
  prep_weight_planes<<<prep_blocks, 256, 0, stream>>>(Wk, wkHi, wkLo, C_, C_);
  prep_weight_planes<<<prep_blocks, 256, 0, stream>>>(Wv, wvHi, wvLo, C_, C_);
  prep_weight_planes<<<prep_blocks, 256, 0, stream>>>(Wproj, wpHi, wpLo, C_, C_);

  const dim3 gemm_grid((B_ * L_) / BM, C_ / BN);       // 128 x 16

  // 2) Projections (q,k with fused per-64 L2 normalization; v raw).
  gemm_bf16split<<<gemm_grid, 256, 0, stream>>>(q, wqHi, wqLo, qn, nullptr, C_, C_, 1);
  gemm_bf16split<<<gemm_grid, 256, 0, stream>>>(k, wkHi, wkLo, kn, nullptr, C_, C_, 1);
  gemm_bf16split<<<gemm_grid, 256, 0, stream>>>(v, wvHi, wvLo, vh, nullptr, C_, C_, 0);

  // 3) attn = kn^T @ vh per head -> packed hi/lo planes.
  attn_kernel<<<dim3(B_ * H_), 128, 0, stream>>>(kn, vh, attnHi, attnLo);

  // 4) x = normalize(0.5 vh + (1/pi) qn@attn) + depthwise conv(vh) -> y.
  combine_kernel<<<dim3(L_ / 128, B_ * H_), 256, 0, stream>>>(
      qn, vh, attnHi, attnLo, Wd, y);

  // 5) out = y @ Wproj + bproj.
  gemm_bf16split<<<gemm_grid, 256, 0, stream>>>(y, wpHi, wpLo, out, bproj, C_, C_, 0);
}
